// MoELayer_20761871908984
// MI455X (gfx1250) — matrix operand order repaired and checked by reading, not compiled, not run on hardware
//
#include <hip/hip_runtime.h>

typedef __bf16 bf16_t;
typedef __bf16 v16bf __attribute__((ext_vector_type(16)));
typedef __bf16 v8bf  __attribute__((ext_vector_type(8)));
typedef float  v8f   __attribute__((ext_vector_type(8)));
typedef float  v4f   __attribute__((ext_vector_type(4)));

#define B_    4
#define S_    1024
#define D_    1024
#define H_    4096
#define E_    8
#define K_    2
#define NTOK  (B_*S_)          // 4096 tokens
#define TILES_PER_E (NTOK/16)  // 256
#define XT_STRIDE 1032         // 1024 + 8 halves pad -> bank-conflict-free A loads
#define HC_   256              // h-chunk width (2 x 16 cols per wave)
#define HB_STRIDE 264          // 256 + 8 halves pad (132 dwords: m*132 % 64 = 4m)

// ---------------- routing ----------------
__global__ void k_init_counts(int* counts) {
    if (threadIdx.x < E_) counts[threadIdx.x] = 0;
}

__global__ void k_route(const int* __restrict__ assign, int* __restrict__ counts,
                        int* __restrict__ lists) {
    int t = blockIdx.x * blockDim.x + threadIdx.x;
    if (t >= NTOK) return;
    int e0 = assign[t * K_ + 0];
    int e1 = assign[t * K_ + 1];
    int p = atomicAdd(&counts[e0], 1);
    lists[e0 * NTOK + p] = t * 2 + 0;            // entry = token*2 + slot
    if (e1 != e0) {
        p = atomicAdd(&counts[e1], 1);
        lists[e1 * NTOK + p] = t * 2 + 1;
    }
}

// ---------------- fp32 -> bf16 (x, elementwise) ----------------
__global__ void k_cvt_x(const float* __restrict__ src, bf16_t* __restrict__ dst) {
    int idx = (blockIdx.x * blockDim.x + threadIdx.x) * 8;
    v4f a = *(const v4f*)(src + idx);
    v4f b = *(const v4f*)(src + idx + 4);
    v8bf o;
    o[0] = (bf16_t)a.x; o[1] = (bf16_t)a.y; o[2] = (bf16_t)a.z; o[3] = (bf16_t)a.w;
    o[4] = (bf16_t)b.x; o[5] = (bf16_t)b.y; o[6] = (bf16_t)b.z; o[7] = (bf16_t)b.w;
    *(v8bf*)(dst + idx) = o;
}

// ---------------- fp32 -> bf16 transposed weights ----------------
// src: [E][R][C] f32 ; dst: [E][C][R] bf16  (R,C multiples of 32)
__global__ void k_cvt_transpose(const float* __restrict__ src, bf16_t* __restrict__ dst,
                                int R, int C) {
    __shared__ float tile[32][33];
    const int e = blockIdx.z;
    const float* s = src + (size_t)e * R * C;
    bf16_t* d = dst + (size_t)e * R * C;
    const int c0 = blockIdx.x * 32, r0 = blockIdx.y * 32;
#pragma unroll
    for (int yy = 0; yy < 4; ++yy) {
        int r = r0 + threadIdx.y + yy * 8;
        tile[threadIdx.y + yy * 8][threadIdx.x] = s[(size_t)r * C + c0 + threadIdx.x];
    }
    __syncthreads();
#pragma unroll
    for (int yy = 0; yy < 4; ++yy) {
        int c = c0 + threadIdx.y + yy * 8;
        d[(size_t)c * R + r0 + threadIdx.x] = (bf16_t)tile[threadIdx.x][threadIdx.y + yy * 8];
    }
}

// ---------------- fused expert MLP (two GEMMs) ----------------
// one workgroup = one expert x one 16-token tile; 8 waves
__global__ __launch_bounds__(256) void k_moe_fused(
    const bf16_t* __restrict__ xb,     // [NTOK][D] bf16
    const bf16_t* __restrict__ W1t,    // [E][H][D] bf16 (transposed)
    const float* __restrict__ b1,      // [E][H]
    const bf16_t* __restrict__ W2t,    // [E][D][H] bf16 (transposed)
    const float* __restrict__ b2,      // [E][D]
    const int* __restrict__ counts,
    const int* __restrict__ lists,
    float* __restrict__ contrib)       // [2][NTOK][D] f32
{
    __shared__ __attribute__((aligned(16))) bf16_t xt[16 * XT_STRIDE];
    __shared__ __attribute__((aligned(16))) bf16_t hbuf[16 * HB_STRIDE];
    __shared__ int rowInfo[16];

    const int e    = blockIdx.x >> 8;
    const int tile = blockIdx.x & 255;
    const int cnt  = counts[e];
    if (tile * 16 >= cnt) return;       // uniform exit

    const int tid = threadIdx.x;
    if (tid < 16) {
        int idx = tile * 16 + tid;
        rowInfo[tid] = (idx < cnt) ? lists[e * NTOK + idx] : -1;
    }
    __syncthreads();

    // ---- stage x tile (16 x D, bf16) into LDS via async global->LDS copies ----
    {
        const int r = tid >> 4, l16 = tid & 15;
        int ri = rowInfo[r];
        int t = (ri >= 0) ? (ri >> 1) : (rowInfo[0] >> 1);   // pad rows replicate row 0
        const bf16_t* xr = xb + (size_t)t * D_;
        const unsigned ldsBase = (unsigned)(uintptr_t)(&xt[0]);
#pragma unroll
        for (int i = 0; i < 8; ++i) {
            int col = l16 * 64 + i * 8;                       // 8 bf16 = 16 B per copy
            unsigned ldsOff = ldsBase + (unsigned)((r * XT_STRIDE + col) * 2);
            unsigned long long ga = (unsigned long long)(uintptr_t)(xr + col);
            asm volatile("global_load_async_to_lds_b128 %0, %1, off"
                         :: "v"(ldsOff), "v"(ga) : "memory");
        }
        asm volatile("s_wait_asynccnt 0" ::: "memory");
    }
    __syncthreads();

    const int wave  = tid >> 5;
    const int lane  = tid & 31;
    const int n16   = lane & 15;
    const int khalf = lane >> 4;
    const int n0    = wave * 128;       // GEMM2 output-column base for this wave

    const bf16_t* W1e = W1t + (size_t)e * H_ * D_;
    const bf16_t* W2e = W2t + (size_t)e * D_ * H_;
    const float*  b1e = b1 + (size_t)e * H_;
    const float*  b2e = b2 + (size_t)e * D_;

    v8f c[8];
#pragma unroll
    for (int j = 0; j < 8; ++j)
        c[j] = (v8f){0.f, 0.f, 0.f, 0.f, 0.f, 0.f, 0.f, 0.f};

    for (int hc = 0; hc < H_; hc += HC_) {
        // ---- GEMM1: two h-column tiles per wave, shared A fragment ----
        const int hcol0 = hc + wave * 32 + n16;        // first tile's B column
        const int hcol1 = hcol0 + 16;                  // second tile's B column
        const bf16_t* bp0 = W1e + (size_t)hcol0 * D_ + khalf * 16;
        const bf16_t* bp1 = W1e + (size_t)hcol1 * D_ + khalf * 16;
        v8f acc0 = (v8f){0.f, 0.f, 0.f, 0.f, 0.f, 0.f, 0.f, 0.f};
        v8f acc1 = (v8f){0.f, 0.f, 0.f, 0.f, 0.f, 0.f, 0.f, 0.f};
#pragma unroll 4
        for (int kk = 0; kk < D_; kk += 32) {
            const bf16_t* ap = &xt[n16 * XT_STRIDE + kk + khalf * 16];
            v8bf alo = *(const v8bf*)ap;
            v8bf ahi = *(const v8bf*)(ap + 8);
            v16bf a = __builtin_shufflevector(alo, ahi,
                        0,1,2,3,4,5,6,7,8,9,10,11,12,13,14,15);
            v16bf b0 = *(const v16bf*)(bp0 + kk);
            v16bf b1v = *(const v16bf*)(bp1 + kk);
            acc0 = __builtin_amdgcn_wmma_f32_16x16x32_bf16(
                       false, a, false, b0, (short)0, acc0, false, false);
            acc1 = __builtin_amdgcn_wmma_f32_16x16x32_bf16(
                       false, a, false, b1v, (short)0, acc1, false, false);
        }
        // bias + relu, write bf16 h-chunk (16 x 256) to LDS
        {
            float bv0 = b1e[hcol0];
            float bv1 = b1e[hcol1];
#pragma unroll
            for (int i = 0; i < 8; ++i) {
                int m = i + khalf * 8;
                float v0 = acc0[i] + bv0; v0 = v0 > 0.f ? v0 : 0.f;
                float v1 = acc1[i] + bv1; v1 = v1 > 0.f ? v1 : 0.f;
                hbuf[m * HB_STRIDE + wave * 32 + n16]      = (bf16_t)v0;
                hbuf[m * HB_STRIDE + wave * 32 + 16 + n16] = (bf16_t)v1;
            }
        }
        __syncthreads();

        // ---- GEMM2: out[:, n0..n0+127] += h_chunk @ W2[hc..hc+256, :] ----
#pragma unroll
        for (int k2 = 0; k2 < HC_; k2 += 32) {
            const bf16_t* ap = &hbuf[n16 * HB_STRIDE + k2 + khalf * 16];
            v8bf alo = *(const v8bf*)ap;
            v8bf ahi = *(const v8bf*)(ap + 8);
            v16bf a = __builtin_shufflevector(alo, ahi,
                        0,1,2,3,4,5,6,7,8,9,10,11,12,13,14,15);
#pragma unroll
            for (int j = 0; j < 8; ++j) {
                const bf16_t* bp = W2e + (size_t)(n0 + j * 16 + n16) * H_
                                       + hc + k2 + khalf * 16;
                v16bf b = *(const v16bf*)bp;
                c[j] = __builtin_amdgcn_wmma_f32_16x16x32_bf16(
                           false, a, false, b, (short)0, c[j], false, false);
            }
        }
        __syncthreads();   // protect hbuf WAR for next chunk
    }

    // ---- write per-(token,slot) contributions (+b2), deterministic, no atomics ----
#pragma unroll
    for (int j = 0; j < 8; ++j) {
        int col = n0 + j * 16 + n16;
        float bb = b2e[col];
#pragma unroll
        for (int i = 0; i < 8; ++i) {
            int m = i + khalf * 8;
            int ri = rowInfo[m];
            if (ri >= 0) {
                int t = ri >> 1, slot = ri & 1;
                contrib[(size_t)slot * NTOK * D_ + (size_t)t * D_ + col] = c[j][i] + bb;
            }
        }
    }
}

// ---------------- final combine: out = (c0 + (e1!=e0)*c1) / k ----------------
__global__ void k_combine(const float* __restrict__ contrib,
                          const int* __restrict__ assign,
                          const int* __restrict__ kptr,
                          float* __restrict__ out) {
    int idx = blockIdx.x * blockDim.x + threadIdx.x;   // over NTOK * D/4
    int t  = idx / (D_ / 4);
    int c4 = idx % (D_ / 4);
    float invk = 1.0f / (float)kptr[0];
    int e0 = assign[t * 2 + 0], e1 = assign[t * 2 + 1];
    v4f v = *((const v4f*)(contrib + (size_t)t * D_) + c4);
    if (e1 != e0) {
        v4f w = *((const v4f*)(contrib + (size_t)NTOK * D_ + (size_t)t * D_) + c4);
        v = v + w;
    }
    v = v * invk;
    *((v4f*)(out + (size_t)t * D_) + c4) = v;
}

// ---------------- launcher ----------------
extern "C" void kernel_launch(void* const* d_in, const int* in_sizes, int n_in,
                              void* d_out, int out_size, void* d_ws, size_t ws_size,
                              hipStream_t stream) {
    const float* x    = (const float*)d_in[0];
    const float* W1   = (const float*)d_in[1];
    const float* b1   = (const float*)d_in[2];
    const float* W2   = (const float*)d_in[3];
    const float* b2   = (const float*)d_in[4];
    const int*  assign = (const int*)d_in[5];
    const int*  kptr   = (const int*)d_in[6];
    float* out = (float*)d_out;

    char* ws = (char*)d_ws;
    // workspace layout
    const size_t OFF_COUNTS  = 0;                                   // 32 B
    const size_t OFF_LIST    = 1024;                                // 128 KB
    const size_t OFF_XB      = OFF_LIST + (size_t)E_ * NTOK * 4;    // 8 MB
    const size_t OFF_W1T     = OFF_XB + (size_t)NTOK * D_ * 2;      // 64 MB
    const size_t OFF_W2T     = OFF_W1T + (size_t)E_ * D_ * H_ * 2;  // 64 MB
    const size_t OFF_CONTRIB = OFF_W2T + (size_t)E_ * D_ * H_ * 2;  // 32 MB

    int*    counts  = (int*)(ws + OFF_COUNTS);
    int*    lists   = (int*)(ws + OFF_LIST);
    bf16_t* xbuf    = (bf16_t*)(ws + OFF_XB);
    bf16_t* W1t     = (bf16_t*)(ws + OFF_W1T);
    bf16_t* W2t     = (bf16_t*)(ws + OFF_W2T);
    float*  contrib = (float*)(ws + OFF_CONTRIB);

    k_init_counts<<<1, 32, 0, stream>>>(counts);
    k_route<<<NTOK / 256, 256, 0, stream>>>(assign, counts, lists);

    k_cvt_x<<<(NTOK * D_ / 8) / 256, 256, 0, stream>>>(x, xbuf);

    dim3 tb(32, 8, 1);
    k_cvt_transpose<<<dim3(H_ / 32, D_ / 32, E_), tb, 0, stream>>>(W1, W1t, D_, H_);
    k_cvt_transpose<<<dim3(D_ / 32, H_ / 32, E_), tb, 0, stream>>>(W2, W2t, H_, D_);

    k_moe_fused<<<E_ * TILES_PER_E, 256, 0, stream>>>(
        xbuf, W1t, b1, W2t, b2, counts, lists, contrib);

    k_combine<<<(NTOK * (D_ / 4)) / 256, 256, 0, stream>>>(contrib, assign, kptr, out);
}
